// ProductKeyRouter_90984587198540
// MI455X (gfx1250) — matrix operand order repaired and checked by reading, not compiled, not run on hardware
//
#include <hip/hip_runtime.h>
#include <hip/hip_bf16.h>

// ---------------- problem constants ----------------
constexpr int NHEAD = 8;
constexpr int QDIM  = 256;
constexpr int SUBD  = 128;   // QDIM/2
constexpr int NSIDE = 512;
constexpr int NK    = 16;
constexpr int DIN   = 1024;
constexpr int NQO   = 2048;  // NHEAD*QDIM
constexpr int CTOK  = 4 * 2048; // B*S

// ---------------- types ----------------
typedef __attribute__((ext_vector_type(16))) __bf16 v16bf;
typedef __attribute__((ext_vector_type(8)))  float  v8f;
typedef unsigned int u32x4 __attribute__((ext_vector_type(4)));
typedef int i32x4 __attribute__((ext_vector_type(4)));
typedef int i32x8 __attribute__((ext_vector_type(8)));

union ABFrag { u32x4 u[2]; v16bf v; };

__device__ __forceinline__ unsigned short f2bf(float f) {
    unsigned u = __float_as_uint(f);
    u += 0x7FFFu + ((u >> 16) & 1u);          // round-to-nearest-even
    return (unsigned short)(u >> 16);
}
__device__ __forceinline__ float bf2f(unsigned short h) {
    return __uint_as_float(((unsigned)h) << 16);
}

#if defined(__has_builtin)
#if __has_builtin(__builtin_amdgcn_tensor_load_to_lds)
#define HAVE_TDM 1
#endif
#endif

#ifdef HAVE_TDM
// Issue a 2D TDM tile load: global (row-major, rowStride elements of bf16)
// -> LDS (compact tile_k x tile_rows). Wave-uniform args only.
__device__ __forceinline__ void tdm_load_2d(unsigned ldsOff, const void* gaddr,
                                            unsigned tile_k, unsigned tile_rows,
                                            unsigned rowStride) {
    const unsigned long long ga = (unsigned long long)gaddr;
    u32x4 g0;
    g0[0] = 1u;                                    // count=1, is_restore=0
    g0[1] = ldsOff;                                // lds_addr (bytes)
    g0[2] = (unsigned)(ga & 0xFFFFFFFFu);          // global_addr[31:0]
    g0[3] = (unsigned)((ga >> 32) & 0x1FFFFFFu) |  // global_addr[56:32]
            (2u << 30);                            // type=2 ("image")
    const unsigned td0 = 0x40000000u;              // huge tensor dims: no OOB clip
    const unsigned td1 = 0x40000000u;
    i32x8 g1;
    g1[0] = (int)(1u << 16);                                   // data_size=1 (2B)
    g1[1] = (int)((td0 & 0xFFFFu) << 16);                      // tensor_dim0 lo
    g1[2] = (int)(((td0 >> 16) & 0xFFFFu) | ((td1 & 0xFFFFu) << 16));
    g1[3] = (int)(((td1 >> 16) & 0xFFFFu) | (tile_k << 16));   // tile_dim0
    g1[4] = (int)(tile_rows & 0xFFFFu);                        // tile_dim1, tile_dim2=0
    g1[5] = (int)rowStride;                                    // tensor_dim0_stride lo
    g1[6] = 0;
    g1[7] = 0;
    i32x4 z = {0, 0, 0, 0};
#if __clang_major__ >= 23
    i32x8 z8 = {0, 0, 0, 0, 0, 0, 0, 0};
    __builtin_amdgcn_tensor_load_to_lds(g0, g1, z, z, z8, 0);
#else
    __builtin_amdgcn_tensor_load_to_lds(g0, g1, z, z, 0);
#endif
}
#endif

// ---------------- conversion kernels ----------------
__global__ void cvt_f32_to_bf16(const float* __restrict__ in,
                                unsigned short* __restrict__ out, int n) {
    int i = blockIdx.x * blockDim.x + threadIdx.x;
    if (i < n) out[i] = f2bf(in[i]);
}

// Wq [DIN][NQO] f32 -> WqT [NQO][DIN] bf16 (transpose so B-operand rows are contiguous-K)
__global__ void cvt_transpose_wq(const float* __restrict__ Wq,
                                 unsigned short* __restrict__ WqT) {
    int i = blockIdx.x * blockDim.x + threadIdx.x;   // over DIN*NQO
    int k = i >> 11;          // / NQO
    int n = i & (NQO - 1);
    WqT[n * DIN + k] = f2bf(Wq[i]);
}

// ---------------- query projection GEMM: Q = Xb @ WqT^T + bq ----------------
// Block: 8 waves, tile M=128 x N=64, K staged 64 at a time through LDS with
// TDM (tensor_load_to_lds) double buffering. Wave w computes rows [w*16, w*16+16).
constexpr int KSTG = 64;                 // K elements per stage
constexpr int NSTG = DIN / KSTG;         // 16 stages

__global__ __launch_bounds__(256) void qproj_gemm(
    const unsigned short* __restrict__ Xb,   // [CTOK][DIN] bf16
    const unsigned short* __restrict__ WqT,  // [NQO][DIN] bf16
    const float*          __restrict__ bq,   // [NQO]
    unsigned short*       __restrict__ Qb)   // [CTOK][NQO] bf16
{
    __shared__ __align__(16) unsigned short ldsA[2][128 * KSTG];  // 2 x 16 KB
    __shared__ __align__(16) unsigned short ldsB[2][64 * KSTG];   // 2 x 8 KB

    const int lane = threadIdx.x & 31;
    const int wave = threadIdx.x >> 5;
    const int mblk = blockIdx.x >> 5;        // NQO/64 == 32 n-blocks
    const int nblk = blockIdx.x & 31;
    const int m0 = mblk * 128;
    const int n0 = nblk * 64;

#ifdef HAVE_TDM
    // prologue: stage 0 into buffer 0 (wave 0 drives the TDM)
    if (wave == 0) {
        tdm_load_2d((unsigned)(unsigned long long)&ldsA[0][0],
                    Xb + (size_t)m0 * DIN, KSTG, 128, DIN);
        tdm_load_2d((unsigned)(unsigned long long)&ldsB[0][0],
                    WqT + (size_t)n0 * DIN, KSTG, 64, DIN);
    }
#else
    // fallback: cooperative copy of stage 0
    for (int i = threadIdx.x; i < 128 * KSTG / 8; i += 256) {
        const int r = i / (KSTG / 8), c = (i % (KSTG / 8)) * 8;
        *(u32x4*)&ldsA[0][r * KSTG + c] =
            *(const u32x4*)(Xb + (size_t)(m0 + r) * DIN + c);
    }
    for (int i = threadIdx.x; i < 64 * KSTG / 8; i += 256) {
        const int r = i / (KSTG / 8), c = (i % (KSTG / 8)) * 8;
        *(u32x4*)&ldsB[0][r * KSTG + c] =
            *(const u32x4*)(WqT + (size_t)(n0 + r) * DIN + c);
    }
#endif

    const int aOff0 = (lane >> 4) * 8;   // A K-chunks {0,16}/{8,24}
    const int bOff  = (lane >> 4) * 8;   // B K-chunks {0,16}/{8,24}
    const int arow  = wave * 16 + (lane & 15);

    v8f acc[4] = {{}, {}, {}, {}};

    for (int ks = 0; ks < NSTG; ++ks) {
        const int buf = ks & 1;
#ifdef HAVE_TDM
        if (wave == 0) {
            if (ks < NSTG - 1) {
                const int ko = (ks + 1) * KSTG;
                tdm_load_2d((unsigned)(unsigned long long)&ldsA[buf ^ 1][0],
                            Xb + (size_t)m0 * DIN + ko, KSTG, 128, DIN);
                tdm_load_2d((unsigned)(unsigned long long)&ldsB[buf ^ 1][0],
                            WqT + (size_t)n0 * DIN + ko, KSTG, 64, DIN);
                __builtin_amdgcn_s_wait_tensorcnt((short)2);  // current buf done
            } else {
                __builtin_amdgcn_s_wait_tensorcnt((short)0);
            }
        }
        __syncthreads();   // buf ready for everyone
#else
        __syncthreads();
#endif
        // compute on buf: 2 k-chunks x 4 n-subtiles
#pragma unroll
        for (int kk = 0; kk < 2; ++kk) {
            ABFrag af;
            const unsigned short* ap = &ldsA[buf][arow * KSTG + kk * 32 + aOff0];
            af.u[0] = *reinterpret_cast<const u32x4*>(ap);
            af.u[1] = *reinterpret_cast<const u32x4*>(ap + 16);
#pragma unroll
            for (int s = 0; s < 4; ++s) {
                ABFrag bf;
                const unsigned short* bp =
                    &ldsB[buf][(s * 16 + (lane & 15)) * KSTG + kk * 32 + bOff];
                bf.u[0] = *reinterpret_cast<const u32x4*>(bp);
                bf.u[1] = *reinterpret_cast<const u32x4*>(bp + 16);
                acc[s] = __builtin_amdgcn_wmma_f32_16x16x32_bf16(
                    false, af.v, false, bf.v, (short)0, acc[s], false, false);
            }
        }
        __syncthreads();   // all reads of buf done before it is overwritten
#ifndef HAVE_TDM
        if (ks < NSTG - 1) {
            const int ko = (ks + 1) * KSTG, nb = buf ^ 1;
            for (int i = threadIdx.x; i < 128 * KSTG / 8; i += 256) {
                const int r = i / (KSTG / 8), c = (i % (KSTG / 8)) * 8;
                *(u32x4*)&ldsA[nb][r * KSTG + c] =
                    *(const u32x4*)(Xb + (size_t)(m0 + r) * DIN + ko + c);
            }
            for (int i = threadIdx.x; i < 64 * KSTG / 8; i += 256) {
                const int r = i / (KSTG / 8), c = (i % (KSTG / 8)) * 8;
                *(u32x4*)&ldsB[nb][r * KSTG + c] =
                    *(const u32x4*)(WqT + (size_t)(n0 + r) * DIN + ko + c);
            }
        }
#endif
    }

    // C/D layout: VGPR r, lanes 0-15 -> M=r, lanes 16-31 -> M=8+r; N = lane&15
    const int mrow = m0 + wave * 16 + (lane >> 4) * 8;
    const int ncol = lane & 15;
#pragma unroll
    for (int s = 0; s < 4; ++s) {
        const float bias = bq[n0 + s * 16 + ncol];
#pragma unroll
        for (int r = 0; r < 8; ++r) {
            Qb[(size_t)(mrow + r) * NQO + n0 + s * 16 + ncol] =
                f2bf(acc[s][r] + bias);
        }
    }
}

// ---------------- fused scores + product-key top-k ----------------
// block: 256 threads (8 waves), 16 tokens x 1 head.
__global__ __launch_bounds__(256) void router_topk(
    const unsigned short* __restrict__ Qb,   // [CTOK][NQO] bf16
    const unsigned short* __restrict__ Kb,   // [NHEAD][2][NSIDE][SUBD] bf16
    int*   __restrict__ outIdx,              // [CTOK][NHEAD][NK]
    float* __restrict__ outScr)              // [CTOK][NHEAD][NK]
{
    __shared__ unsigned short s_sc[2][16][NSIDE];   // 32 KB score tile
    __shared__ float s_tv[16][2][NK];
    __shared__ int   s_ti[16][2][NK];

    const int lane = threadIdx.x & 31;
    const int wave = threadIdx.x >> 5;
    const int tok0 = blockIdx.x * 16;
    const int h    = blockIdx.y;

    const int hf    = wave >> 2;
    const int tbase = (wave & 3) * 8;

    ABFrag a[4];
    const int qrow  = tok0 + (lane & 15);
    const int qcol0 = h * QDIM + hf * SUBD;
    const int aOff0 = (lane >> 4) * 8;
#pragma unroll
    for (int kk = 0; kk < 4; ++kk) {
        const unsigned short* ap = Qb + (size_t)qrow * NQO + qcol0 + kk * 32 + aOff0;
        a[kk].u[0] = *reinterpret_cast<const u32x4*>(ap);
        a[kk].u[1] = *reinterpret_cast<const u32x4*>(ap + 16);
    }

    const unsigned short* keyBase = Kb + (size_t)((h * 2 + hf) * NSIDE) * SUBD;
    const int bOff = (lane >> 4) * 8;
    const int mrow = (lane >> 4) * 8;
    const int ncol = lane & 15;

    const unsigned short* keyLane =
        keyBase + (size_t)(tbase * 16 + (lane & 15)) * SUBD + bOff;

    ABFrag b[2][4];
#pragma unroll
    for (int kk = 0; kk < 4; ++kk) {
        b[0][kk].u[0] = *reinterpret_cast<const u32x4*>(keyLane + kk * 32);
        b[0][kk].u[1] = *reinterpret_cast<const u32x4*>(keyLane + kk * 32 + 16);
    }

    for (int j = 0; j < 8; ++j) {
        const int cur = j & 1;
        const int nxt = cur ^ 1;
        if (j < 7) {
            const unsigned short* kp = keyLane + (size_t)(j + 1) * 16 * SUBD;
#pragma unroll
            for (int kk = 0; kk < 4; ++kk) {
                b[nxt][kk].u[0] = *reinterpret_cast<const u32x4*>(kp + kk * 32);
                b[nxt][kk].u[1] = *reinterpret_cast<const u32x4*>(kp + kk * 32 + 16);
            }
        }
        if (j < 6)
            __builtin_prefetch(keyLane + (size_t)(j + 2) * 16 * SUBD, 0, 0);

        v8f c = {};
#pragma unroll
        for (int kk = 0; kk < 4; ++kk) {
            c = __builtin_amdgcn_wmma_f32_16x16x32_bf16(
                false, a[kk].v, false, b[cur][kk].v, (short)0, c, false, false);
        }

        const int e0 = (tbase + j) * 16;
#pragma unroll
        for (int r = 0; r < 8; ++r)
            s_sc[hf][mrow + r][e0 + ncol] = f2bf(c[r]);
    }

    __syncthreads();

    const unsigned short NEG_INF_BF = 0xFF80;
    const float NEG_INF = -__builtin_inff();

    for (int tt = 0; tt < 2; ++tt) {
        const int tl = wave * 2 + tt;

        for (int s = 0; s < 2; ++s) {
            for (int k = 0; k < NK; ++k) {
                float best = NEG_INF; int bidx = 0x7FFFFFFF;
#pragma unroll
                for (int i = 0; i < NSIDE / 32; ++i) {
                    const int e = lane + i * 32;
                    const float v = bf2f(s_sc[s][tl][e]);
                    if (v > best) { best = v; bidx = e; }
                }
#pragma unroll
                for (int off = 16; off > 0; off >>= 1) {
                    const float ov = __shfl_xor(best, off, 32);
                    const int   oi = __shfl_xor(bidx, off, 32);
                    if (ov > best || (ov == best && oi < bidx)) { best = ov; bidx = oi; }
                }
                if (lane == 0) { s_tv[tl][s][k] = best; s_ti[tl][s][k] = bidx; }
                if ((bidx & 31) == lane) s_sc[s][tl][bidx] = NEG_INF_BF;
            }
        }

        float cv[8];
#pragma unroll
        for (int sl = 0; sl < 8; ++sl) {
            const int f = lane + 32 * sl;
            cv[sl] = s_tv[tl][0][f >> 4] + s_tv[tl][1][f & 15];
        }

        const int ctok = tok0 + tl;
        int*   oI = outIdx + ((size_t)ctok * NHEAD + h) * NK;
        float* oS = outScr + ((size_t)ctok * NHEAD + h) * NK;

        for (int k = 0; k < NK; ++k) {
            float best = NEG_INF; int bf_ = 0x7FFFFFFF;
#pragma unroll
            for (int sl = 0; sl < 8; ++sl) {
                const int f = lane + 32 * sl;
                if (cv[sl] > best) { best = cv[sl]; bf_ = f; }
            }
#pragma unroll
            for (int off = 16; off > 0; off >>= 1) {
                const float ov = __shfl_xor(best, off, 32);
                const int   oi = __shfl_xor(bf_, off, 32);
                if (ov > best || (ov == best && oi < bf_)) { best = ov; bf_ = oi; }
            }
            const int owner = bf_ & 31, slot = bf_ >> 5;
#pragma unroll
            for (int sl = 0; sl < 8; ++sl)
                if (owner == lane && sl == slot) cv[sl] = NEG_INF;
            if (lane == 0) {
                const int i_ = bf_ >> 4, j_ = bf_ & 15;
                oI[k] = s_ti[tl][0][i_] * NSIDE + s_ti[tl][1][j_];
                oS[k] = best;
            }
        }
    }
}

// ---------------- launcher ----------------
extern "C" void kernel_launch(void* const* d_in, const int* in_sizes, int n_in,
                              void* d_out, int out_size, void* d_ws, size_t ws_size,
                              hipStream_t stream) {
    const float* x    = (const float*)d_in[0];   // [B,S,DIN]
    const float* Wq   = (const float*)d_in[1];   // [DIN, NQO]
    const float* bq   = (const float*)d_in[2];   // [NQO]
    const float* keys = (const float*)d_in[3];   // [NHEAD,2,NSIDE,SUBD]

    char* ws = (char*)d_ws;
    unsigned short* Xb  = (unsigned short*)ws; ws += (size_t)CTOK * DIN * 2;
    unsigned short* WqT = (unsigned short*)ws; ws += (size_t)NQO  * DIN * 2;
    unsigned short* Kb  = (unsigned short*)ws; ws += (size_t)NHEAD * 2 * NSIDE * SUBD * 2;
    unsigned short* Qb  = (unsigned short*)ws; ws += (size_t)CTOK * NQO * 2;

    const int nx = CTOK * DIN;                 // 8M
    cvt_f32_to_bf16<<<nx / 256, 256, 0, stream>>>(x, Xb, nx);
    cvt_transpose_wq<<<(DIN * NQO) / 256, 256, 0, stream>>>(Wq, WqT);
    const int nk = NHEAD * 2 * NSIDE * SUBD;   // 1M
    cvt_f32_to_bf16<<<nk / 256, 256, 0, stream>>>(keys, Kb, nk);

    // (CTOK/128) m-blocks x (NQO/64) n-blocks
    qproj_gemm<<<(CTOK / 128) * (NQO / 64), 256, 0, stream>>>(Xb, WqT, bq, Qb);

    int*   outIdx = (int*)d_out;                                   // indices first
    float* outScr = (float*)d_out + (size_t)CTOK * NHEAD * NK;     // then scores
    dim3 grid(CTOK / 16, NHEAD);
    router_topk<<<grid, 256, 0, stream>>>(Qb, Kb, outIdx, outScr);
}
